// APPNPNodeRegressor_68659347194260
// MI455X (gfx1250) — matrix operand order repaired and checked by reading, not compiled, not run on hardware
//
#include <hip/hip_runtime.h>

#define HID 128
#define ALPHA 0.1f
#define KSTEPS 10

typedef __attribute__((ext_vector_type(16))) _Float16 v16h;
typedef __attribute__((ext_vector_type(8)))  float    v8f;

// ---------------------------------------------------------------------------
// hardware f32 scatter-add: guarantee GLOBAL_ATOMIC_ADD_F32 (no CAS loop).
// Device scope so the RMW resolves at the L2 atomic units (cross-WGP).
// No-return form -> tracked by STOREcnt; S_ENDPGM's implicit wait covers it.
// ---------------------------------------------------------------------------
__device__ __forceinline__ void gfadd(float* p, float v) {
  asm volatile("global_atomic_add_f32 %0, %1, off scope:SCOPE_DEV"
               :
               : "v"(p), "v"(v)
               : "memory");
}

// ---------------------------------------------------------------------------
// elementwise helpers
// ---------------------------------------------------------------------------
__global__ void fill_f32(float* __restrict__ p, float v, size_t n) {
  size_t i = (size_t)blockIdx.x * blockDim.x + threadIdx.x;
  if (i < n) p[i] = v;
}

__global__ void cvt_f32_to_f16(const float* __restrict__ in, _Float16* __restrict__ out, int n) {
  int i = blockIdx.x * blockDim.x + threadIdx.x;
  if (i < n) out[i] = (_Float16)in[i];
}

__global__ void deg_count(const int* __restrict__ dst, float* __restrict__ deg, int E) {
  int e = blockIdx.x * blockDim.x + threadIdx.x;
  if (e < E) gfadd(&deg[dst[e]], 1.0f);
}

__global__ void rsqrt_inplace(float* __restrict__ p, int n) {
  int i = blockIdx.x * blockDim.x + threadIdx.x;
  if (i < n) p[i] = rsqrtf(p[i]);   // deg >= 1 always (self-loop)
}

// ---------------------------------------------------------------------------
// WMMA MLP GEMM:  out = relu(A @ W^T + b)
//   A: [nRows, KD]  (f32 if AF32, else f16), row-major
//   W: [HID, KD]    f16, row-major
// One block = 8 waves handles a 16-row slab and all 128 output columns
// (one 16x16 C tile per wave). A tile staged through LDS, read in contiguous
// k order (k = 16g + s); B built straight from W (row-major == B^T tile).
// ---------------------------------------------------------------------------
template <int KD, bool AF32>
__global__ __launch_bounds__(256) void mlp_gemm(const void* __restrict__ Aptr,
                                                const _Float16* __restrict__ W,
                                                const float* __restrict__ bias,
                                                _Float16* __restrict__ outH,
                                                float* __restrict__ outF,
                                                float* __restrict__ outF2,
                                                int nRows) {
  __shared__ _Float16 As[16][34];   // pad: 17-dword row stride, conflict-free

  const int tid  = threadIdx.x;
  const int wave = tid >> 5;        // 0..7 -> column tile
  const int lane = tid & 31;
  const int g    = lane >> 4;       // lane group 0/1
  const int nl   = lane & 15;
  const int r0   = blockIdx.x * 16;
  const int ncol = wave * 16 + nl;  // output column 0..127

  v8f c = {};

  for (int k0 = 0; k0 < KD; k0 += 32) {
    __syncthreads();
    {
      // cooperative A tile load: 16 x 32 halfs, 2 elements per thread
      int row  = tid >> 4;          // 0..15
      int col  = (tid & 15) * 2;    // 0,2,...,30
      int arow = r0 + row; if (arow >= nRows) arow = nRows - 1;
      size_t base = (size_t)arow * KD + k0 + col;
      if (AF32) {
        const float* A = (const float*)Aptr;
        As[row][col]     = (_Float16)A[base];
        As[row][col + 1] = (_Float16)A[base + 1];
      } else {
        const _Float16* A = (const _Float16*)Aptr;
        As[row][col]     = A[base];
        As[row][col + 1] = A[base + 1];
      }
    }
    __syncthreads();

    v16h a, b;
    // A layout: VGPR i -> K = 16g + 2i (+j)
#pragma unroll
    for (int i = 0; i < 8; ++i) {
      int kb = 16 * g + 2 * i;
      a[2 * i]     = As[nl][kb];
      a[2 * i + 1] = As[nl][kb + 1];
    }
    // B layout: lane (g,nl) holds column ncol, K = 16g + t  ->  W[ncol][k0+16g+t]
    const _Float16* wrow = W + (size_t)ncol * KD + k0 + 16 * g;
#pragma unroll
    for (int t = 0; t < 16; ++t) b[t] = wrow[t];

    c = __builtin_amdgcn_wmma_f32_16x16x32_f16(false, a, false, b, (short)0, c,
                                               false, false);
  }

  const float bv = bias[ncol];
#pragma unroll
  for (int v = 0; v < 8; ++v) {
    int row = r0 + v + 8 * g;       // C layout: VGPR v -> M = v + 8g, N = nl
    if (row < nRows) {
      float val = c[v] + bv;
      val = val > 0.f ? val : 0.f;
      size_t idx = (size_t)row * HID + ncol;
      if (outH)  outH[idx]  = (_Float16)val;
      if (outF)  outF[idx]  = val;
      if (outF2) outF2[idx] = val;
    }
  }
}

// ---------------------------------------------------------------------------
// propagation: one wave per edge, float4 per lane (128 floats/row)
// ---------------------------------------------------------------------------
__global__ __launch_bounds__(256) void scatter_edges(const int* __restrict__ src,
                                                     const int* __restrict__ dst,
                                                     const float* __restrict__ dinv,
                                                     const float* __restrict__ h,
                                                     float* __restrict__ agg, int E) {
  int gid = blockIdx.x * blockDim.x + threadIdx.x;
  int e   = gid >> 5;
  if (e >= E) return;
  int lane = threadIdx.x & 31;
  int s = src[e], d = dst[e];
  float norm = dinv[s] * dinv[d];
  float4 hv  = ((const float4*)(h + (size_t)s * HID))[lane];
  float* ap  = agg + (size_t)d * HID + lane * 4;
  gfadd(ap + 0, norm * hv.x);
  gfadd(ap + 1, norm * hv.y);
  gfadd(ap + 2, norm * hv.z);
  gfadd(ap + 3, norm * hv.w);
}

// h <- 0.9*(agg + dinv^2 * h) + 0.1*h0 ; agg <- 0  (self-loop folded in here)
__global__ __launch_bounds__(256) void combine_step(float* __restrict__ h,
                                                    float* __restrict__ agg,
                                                    const float* __restrict__ h0,
                                                    const float* __restrict__ dinv, int Nn) {
  int gid  = blockIdx.x * blockDim.x + threadIdx.x;
  int node = gid >> 5;
  if (node >= Nn) return;
  int lane = threadIdx.x & 31;
  float di = dinv[node];
  float sl = di * di;
  size_t base = (size_t)node * HID + lane * 4;
  float4 hv  = *(const float4*)(h + base);
  float4 av  = *(const float4*)(agg + base);
  float4 h0v = *(const float4*)(h0 + base);
  float4 nv;
  nv.x = (1.f - ALPHA) * (av.x + sl * hv.x) + ALPHA * h0v.x;
  nv.y = (1.f - ALPHA) * (av.y + sl * hv.y) + ALPHA * h0v.y;
  nv.z = (1.f - ALPHA) * (av.z + sl * hv.z) + ALPHA * h0v.z;
  nv.w = (1.f - ALPHA) * (av.w + sl * hv.w) + ALPHA * h0v.w;
  *(float4*)(h + base) = nv;
  float4 z = {0.f, 0.f, 0.f, 0.f};
  *(float4*)(agg + base) = z;
}

// out[n] = dot(h[n,:], Wo) + bo   (one wave per node, shuffle reduction)
__global__ __launch_bounds__(256) void out_dot(const float* __restrict__ h,
                                               const float* __restrict__ Wo,
                                               const float* __restrict__ bo,
                                               float* __restrict__ out, int Nn) {
  int gid  = blockIdx.x * blockDim.x + threadIdx.x;
  int node = gid >> 5;
  if (node >= Nn) return;
  int lane = threadIdx.x & 31;
  float4 hv = ((const float4*)(h + (size_t)node * HID))[lane];
  float4 wv = ((const float4*)Wo)[lane];
  float p = hv.x * wv.x + hv.y * wv.y + hv.z * wv.z + hv.w * wv.w;
#pragma unroll
  for (int off = 16; off > 0; off >>= 1) p += __shfl_down(p, off, 32);
  if (lane == 0) out[node] = p + bo[0];
}

// ---------------------------------------------------------------------------
extern "C" void kernel_launch(void* const* d_in, const int* in_sizes, int n_in,
                              void* d_out, int out_size, void* d_ws, size_t ws_size,
                              hipStream_t stream) {
  const float* x  = (const float*)d_in[0];
  const int*   ei = (const int*)d_in[1];
  const float* W1 = (const float*)d_in[2];
  const float* b1 = (const float*)d_in[3];
  const float* W2 = (const float*)d_in[4];
  const float* b2 = (const float*)d_in[5];
  const float* W3 = (const float*)d_in[6];
  const float* b3 = (const float*)d_in[7];
  const float* Wo = (const float*)d_in[8];
  const float* bo = (const float*)d_in[9];
  float* out = (float*)d_out;

  const int IN = 256;
  const int Nn = in_sizes[0] / IN;   // 100000
  const int E  = in_sizes[1] / 2;    // 1600000
  const int* srcI = ei;
  const int* dstI = ei + E;

  // workspace carve-out (256B aligned)
  char* ws = (char*)d_ws;
  size_t off = 0;
  auto alloc = [&](size_t bytes) -> char* {
    char* p = ws + off;
    off += (bytes + 255) & ~(size_t)255;
    return p;
  };
  float*    dinv = (float*)alloc((size_t)Nn * 4);
  float*    h0   = (float*)alloc((size_t)Nn * HID * 4);
  float*    hcur = (float*)alloc((size_t)Nn * HID * 4);
  float*    agg  = (float*)alloc((size_t)Nn * HID * 4);
  _Float16* h1h  = (_Float16*)alloc((size_t)Nn * HID * 2);
  _Float16* h2h  = (_Float16*)alloc((size_t)Nn * HID * 2);
  _Float16* w1h  = (_Float16*)alloc((size_t)HID * IN * 2);
  _Float16* w2h  = (_Float16*)alloc((size_t)HID * HID * 2);
  _Float16* w3h  = (_Float16*)alloc((size_t)HID * HID * 2);

  // weights -> f16
  cvt_f32_to_f16<<<(HID * IN + 255) / 256, 256, 0, stream>>>(W1, w1h, HID * IN);
  cvt_f32_to_f16<<<(HID * HID + 255) / 256, 256, 0, stream>>>(W2, w2h, HID * HID);
  cvt_f32_to_f16<<<(HID * HID + 255) / 256, 256, 0, stream>>>(W3, w3h, HID * HID);

  // degrees (self-loop => init 1.0) -> dinv = rsqrt(deg)
  fill_f32<<<(Nn + 255) / 256, 256, 0, stream>>>(dinv, 1.0f, (size_t)Nn);
  deg_count<<<(E + 255) / 256, 256, 0, stream>>>(dstI, dinv, E);
  rsqrt_inplace<<<(Nn + 255) / 256, 256, 0, stream>>>(dinv, Nn);

  // zero the scatter accumulator
  size_t nh = (size_t)Nn * HID;
  fill_f32<<<(unsigned)((nh + 255) / 256), 256, 0, stream>>>(agg, 0.0f, nh);

  // MLP encoder (WMMA)
  int nb = (Nn + 15) / 16;
  mlp_gemm<256, true ><<<nb, 256, 0, stream>>>(x,   w1h, b1, h1h, nullptr, nullptr, Nn);
  mlp_gemm<128, false><<<nb, 256, 0, stream>>>(h1h, w2h, b2, h2h, nullptr, nullptr, Nn);
  mlp_gemm<128, false><<<nb, 256, 0, stream>>>(h2h, w3h, b3, nullptr, h0, hcur, Nn);

  // APPNP propagation
  unsigned sblocks = (unsigned)(((size_t)E * 32 + 255) / 256);
  unsigned cblocks = (unsigned)(((size_t)Nn * 32 + 255) / 256);
  for (int k = 0; k < KSTEPS; ++k) {
    scatter_edges<<<sblocks, 256, 0, stream>>>(srcI, dstI, dinv, hcur, agg, E);
    combine_step<<<cblocks, 256, 0, stream>>>(hcur, agg, h0, dinv, Nn);
  }

  // readout
  out_dot<<<cblocks, 256, 0, stream>>>(hcur, Wo, bo, out, Nn);
}
